// FamNetSimilarity_51488067945079
// MI455X (gfx1250) — matrix operand order repaired and checked by reading, not compiled, not run on hardware
//
#include <hip/hip_runtime.h>
#include <hip/hip_bf16.h>

typedef __attribute__((ext_vector_type(16))) __bf16 v16bf;
typedef __attribute__((ext_vector_type(8)))  __bf16 v8bf;
typedef __attribute__((ext_vector_type(8)))  float  v8f;

// ---------------- problem constants (from setup_inputs) ----------------
constexpr int Bv = 4, Nv = 3;
constexpr int C3 = 512,  H3 = 128, W3 = 128;
constexpr int C4 = 1024, H4 = 64,  W4 = 64;
// worst-case patch / filter-canvas extents for these boxes (compile-time grids)
constexpr int MAXP3 = 24, MAXF3 = 28;
constexpr int MAXP4 = 16, MAXF4 = 20;
constexpr int MSTR  = 32;  // ints of metadata per (batch, level)
// meta slots: [0..11] t,l,ch,cw per box; [12] mh; [13] mw;
// [14..16] hs per scale; [17..19] ws; [20..22] oy; [23..25] ox; [26] Fh; [27] Fw

// ---------------- workspace layout (bytes, all 1 KiB aligned) ----------------
constexpr size_t OFF_META = 0;                                   // 1024 B
constexpr size_t OFF_F3   = 1024;
constexpr size_t SZ_F3    = (size_t)Bv * H3 * W3 * C3 * 2;       // bf16 NHWC
constexpr size_t OFF_F4   = OFF_F3 + SZ_F3;
constexpr size_t SZ_F4    = (size_t)Bv * H4 * W4 * C4 * 2;
constexpr size_t OFF_P3   = OFF_F4 + SZ_F4;
constexpr size_t SZ_P3    = (size_t)Bv * Nv * MAXP3 * MAXP3 * C3 * 2;
constexpr size_t OFF_P4   = OFF_P3 + SZ_P3;
constexpr size_t SZ_P4    = (size_t)Bv * Nv * MAXP4 * MAXP4 * C4 * 2;
constexpr size_t OFF_FL3  = OFF_P4 + SZ_P4;
constexpr size_t SZ_FL3   = (size_t)Bv * MAXF3 * MAXF3 * 16 * C3 * 2;
constexpr size_t OFF_FL4  = OFF_FL3 + SZ_FL3;
constexpr size_t SZ_FL4   = (size_t)Bv * MAXF4 * MAXF4 * 16 * C4 * 2;
constexpr size_t OFF_S3   = OFF_FL4 + SZ_FL4;
constexpr size_t SZ_S3    = (size_t)Bv * 9 * H3 * W3 * 4;        // f32 sims
constexpr size_t OFF_S4   = OFF_S3 + SZ_S3;

// ---------------- K0: shape metadata from tlbr (double math = numpy f64) ----
__global__ void meta_kernel(const int* __restrict__ tlbr, const int* __restrict__ ihp,
                            const int* __restrict__ iwp, int* __restrict__ meta) {
  int t = threadIdx.x;
  if (t >= Bv * 2) return;
  int b = t >> 1, lev = t & 1;
  int HF = lev ? H4 : H3, WF = lev ? W4 : W3;
  double sh = (double)HF / (double)(*ihp);
  double sw = (double)WF / (double)(*iwp);
  int* mt = meta + (b * 2 + lev) * MSTR;
  int mh = 0, mw = 0;
  for (int n = 0; n < Nv; ++n) {
    const int* bx = tlbr + (b * Nv + n) * 4;
    int tt = (int)fmax(floor((double)bx[0] * sh), 0.0);
    int ll = (int)fmax(floor((double)bx[1] * sw), 0.0);
    int bb = (int)fmin(ceil((double)bx[2] * sh) + 1.0, (double)HF);
    int rr = (int)fmin(ceil((double)bx[3] * sw) + 1.0, (double)WF);
    int chh = bb - tt, cww = rr - ll;
    mt[n * 4 + 0] = tt; mt[n * 4 + 1] = ll; mt[n * 4 + 2] = chh; mt[n * 4 + 3] = cww;
    mh = max(mh, chh); mw = max(mw, cww);
  }
  mt[12] = mh; mt[13] = mw;
  const double sc[3] = {1.0, 0.9, 1.1};
  int Fh = 0, Fw = 0;
  for (int s = 0; s < 3; ++s) {
    int hs, wsz;
    if (s == 0) { hs = mh; wsz = mw; }
    else {
      hs  = (int)ceil((double)mh * sc[s]); if (hs  <= 1) hs  = mh;
      wsz = (int)ceil((double)mw * sc[s]); if (wsz <= 1) wsz = mw;
    }
    mt[14 + s] = hs; mt[17 + s] = wsz;
    Fh = max(Fh, hs); Fw = max(Fw, wsz);
  }
  for (int s = 0; s < 3; ++s) {
    mt[20 + s] = Fh / 2 - mt[14 + s] / 2;   // center-embed offsets: 'same' pads align
    mt[23 + s] = Fw / 2 - mt[17 + s] / 2;
  }
  mt[26] = Fh; mt[27] = Fw;
}

// ---------------- K1: NCHW f32 -> NHWC bf16 (LDS-tiled transpose) ----------
template <int C, int HW>
__global__ __launch_bounds__(1024) void nhwc_kernel(const float* __restrict__ in,
                                                    __bf16* __restrict__ out) {
  __shared__ float tile[32][33];
  int b = blockIdx.z;
  int p0 = blockIdx.x * 32, c0 = blockIdx.y * 32;
  const float* inb = in + (size_t)b * C * HW;
  tile[threadIdx.y][threadIdx.x] =
      inb[(size_t)(c0 + threadIdx.y) * HW + (p0 + threadIdx.x)];
  __syncthreads();
  out[(size_t)b * HW * C + (size_t)(p0 + threadIdx.y) * C + (c0 + threadIdx.x)] =
      (__bf16)tile[threadIdx.x][threadIdx.y];
}

// ---------------- K2: ROI crop -> common-size patch (bilinear #1) ----------
template <int C, int HF, int WF, int MAXP, int LEV>
__global__ __launch_bounds__(256) void patches_kernel(const int* __restrict__ meta,
                                                      const __bf16* __restrict__ feat,
                                                      __bf16* __restrict__ patch) {
  int p = blockIdx.x;
  int px = p % MAXP; p /= MAXP;
  int py = p % MAXP; p /= MAXP;
  int n  = p % Nv;   p /= Nv;
  int b  = p;
  const int* mt = meta + (b * 2 + LEV) * MSTR;
  int mh = mt[12], mw = mt[13];
  if (py >= mh || px >= mw) return;
  int t = mt[n*4+0], l = mt[n*4+1], chh = mt[n*4+2], cww = mt[n*4+3];
  float ys = fminf(fmaxf((py + 0.5f) * ((float)chh / (float)mh) - 0.5f, 0.f), (float)(chh - 1));
  float xs = fminf(fmaxf((px + 0.5f) * ((float)cww / (float)mw) - 0.5f, 0.f), (float)(cww - 1));
  int y0 = (int)floorf(ys), x0 = (int)floorf(xs);
  int y1 = min(y0 + 1, chh - 1), x1 = min(x0 + 1, cww - 1);
  float wy = ys - y0, wx = xs - x0;
  const __bf16* fb = feat + (size_t)b * HF * WF * C;
  size_t r0 = (size_t)(t + y0) * WF * C, r1 = (size_t)(t + y1) * WF * C;
  size_t ca = (size_t)(l + x0) * C,      cb = (size_t)(l + x1) * C;
  __bf16* pp = patch + ((((size_t)b * Nv + n) * MAXP + py) * MAXP + px) * C;
  for (int c = threadIdx.x; c < C; c += 256) {
    float ra = (float)fb[r0 + ca + c] * (1.f - wy) + (float)fb[r1 + ca + c] * wy;
    float rb = (float)fb[r0 + cb + c] * (1.f - wy) + (float)fb[r1 + cb + c] * wy;
    pp[c] = (__bf16)(ra * (1.f - wx) + rb * wx);
  }
}

// ---- K3: patch -> scaled filter, zero-embedded into [Fh][Fw][16][C] canvas ----
// filter slot f = s*3 + n (s=scale idx, n=box); slots 9..15 are zero padding.
template <int C, int MAXP, int MAXF, int LEV>
__global__ __launch_bounds__(256) void filters_kernel(const int* __restrict__ meta,
                                                      const __bf16* __restrict__ patch,
                                                      __bf16* __restrict__ filt) {
  int q = blockIdx.x;
  int f  = q % 16;   q /= 16;
  int fx = q % MAXF; q /= MAXF;
  int fy = q % MAXF; q /= MAXF;
  int b  = q;
  __bf16* out = filt + ((((size_t)b * MAXF + fy) * MAXF + fx) * 16 + f) * C;
  const int* mt = meta + (b * 2 + LEV) * MSTR;
  bool zero = true;
  float wy = 0.f, wx = 0.f;
  int y0 = 0, y1 = 0, x0 = 0, x1 = 0, n = 0;
  if (f < 9) {
    int s = f / 3; n = f % 3;
    int hs = mt[14+s], wsz = mt[17+s], oy = mt[20+s], ox = mt[23+s];
    int ry = fy - oy, rx = fx - ox;
    if (ry >= 0 && ry < hs && rx >= 0 && rx < wsz) {
      zero = false;
      int mh = mt[12], mw = mt[13];
      float ys = fminf(fmaxf((ry + 0.5f) * ((float)mh / (float)hs)  - 0.5f, 0.f), (float)(mh - 1));
      float xs = fminf(fmaxf((rx + 0.5f) * ((float)mw / (float)wsz) - 0.5f, 0.f), (float)(mw - 1));
      y0 = (int)floorf(ys); x0 = (int)floorf(xs);
      y1 = min(y0 + 1, mh - 1); x1 = min(x0 + 1, mw - 1);
      wy = ys - y0; wx = xs - x0;
    }
  }
  if (zero) {
    for (int c = threadIdx.x; c < C; c += 256) out[c] = (__bf16)0.f;
    return;
  }
  const __bf16* pb  = patch + (((size_t)b * Nv + n) * MAXP * MAXP) * C;
  const __bf16* p00 = pb + ((size_t)y0 * MAXP + x0) * C;
  const __bf16* p01 = pb + ((size_t)y0 * MAXP + x1) * C;
  const __bf16* p10 = pb + ((size_t)y1 * MAXP + x0) * C;
  const __bf16* p11 = pb + ((size_t)y1 * MAXP + x1) * C;
  for (int c = threadIdx.x; c < C; c += 256) {
    float ra = (float)p00[c] * (1.f - wy) + (float)p10[c] * wy;
    float rb = (float)p01[c] * (1.f - wy) + (float)p11[c] * wy;
    out[c] = (__bf16)(ra * (1.f - wx) + rb * wx);
  }
}

// ---------------- K4: row-stationary conv GEMM, LDS-staged A ----------------
// One block owns one output row (WAVES waves x 16 px). For each (fy, c-chunk)
// the row's 32-channel slice (+halo) is DMA'd to LDS once via
// global_load_async_to_lds_b128 and reused by every fx tap -> ~Fw x less L2
// A-traffic. B (9-filter canvas, 16-slot padded) streams from L2.
// A layout (16-bit 16x32): lane L -> M=L&15; K = e + (L<16 ? 0 : 16).
// B layout (32x16): lane L -> N=L&15; K = e + (L<16 ? 0 : 16).
template <int C, int HF, int WF, int MAXF, int LEV, int WAVES>
__global__ __launch_bounds__(32 * WAVES) void conv_wmma_kernel(
    const int* __restrict__ meta, const __bf16* __restrict__ feat,
    const __bf16* __restrict__ filt, float* __restrict__ sim) {
  constexpr int NCOLMAX = WF + MAXF;   // row + halo columns
  constexpr int CSTR    = 80;          // bytes/col: 16B aligned, 20-bank skew
  __shared__ __align__(16) char sA[NCOLMAX * CSTR];

  const int tid  = threadIdx.x;
  const int lane = tid & 31;
  const int wave = tid >> 5;
  const int b    = blockIdx.y;
  const int y    = blockIdx.x;

  const int* mt = meta + (b * 2 + LEV) * MSTR;
  const int Fh = mt[26], Fw = mt[27];
  const int hH = Fh >> 1, hW = Fw >> 1;
  const int mrow  = lane & 15;             // A row / B column / D column
  const int xpix  = wave * 16 + mrow;      // this lane's output column
  const int kselA = (lane < 16) ? 0 : 16;
  const int kselB = (lane < 16) ? 0 : 16;
  const __bf16* fb = feat + (size_t)b * HF * WF * C;
  const __bf16* gb = filt + (size_t)b * MAXF * MAXF * 16 * C + (size_t)mrow * C;
  const int ncols  = WF + Fw - 1;
  const int nchunk = ncols * 4;            // 16B chunks, 4 per column

  v8f acc{};
  for (int fy = 0; fy < Fh; ++fy) {
    const int yin = y + fy - hH;
    const bool yok = ((unsigned)yin < (unsigned)HF);
    const __bf16* frow = fb + (size_t)yin * WF * C;
    const __bf16* grow = gb + (size_t)fy * MAXF * 16 * C;
    for (int c0 = 0; c0 < C; c0 += 32) {
      __syncthreads();                     // prior taps finished reading sA
      // --- stage row chunk (+halo) into LDS via async DMA -----------------
      for (int i = tid; i < nchunk; i += 32 * WAVES) {
        const int col = i >> 2, part = i & 3;
        const int xin = col - hW;
        char* dst = sA + col * CSTR + part * 16;
        if (yok && (unsigned)xin < (unsigned)WF) {
          const __bf16* src = frow + (size_t)xin * C + c0 + part * 8;
#if defined(__gfx1250__)
          asm volatile("global_load_async_to_lds_b128 %0, %1, off"
                       :
                       : "v"((unsigned)(uintptr_t)dst), "v"(src)
                       : "memory");
#else
          *(v8bf*)dst = *(const v8bf*)src;
#endif
        } else {
          v8bf z{};
          *(v8bf*)dst = z;                 // zero halo / OOB row
        }
      }
#if defined(__gfx1250__)
      asm volatile("s_wait_asynccnt 0x0" ::: "memory");
#endif
      __syncthreads();
      // --- sweep all fx taps against the staged chunk ---------------------
      #pragma unroll 2
      for (int fx = 0; fx < Fw; ++fx) {
        const char* ac = sA + (size_t)(xpix + fx) * CSTR + kselA * 2;
        const v8bf lo = *(const v8bf*)(ac);        // K = c0+kselA   .. +7
        const v8bf hi = *(const v8bf*)(ac + 16);   // K = c0+kselA+8 .. +15
        v16bf av;
        #pragma unroll
        for (int i2 = 0; i2 < 8; ++i2) { av[i2] = lo[i2]; av[i2 + 8] = hi[i2]; }
        const v16bf bv = *(const v16bf*)(grow + (size_t)fx * 16 * C + c0 + kselB);
        acc = __builtin_amdgcn_wmma_f32_16x16x32_bf16(
            /*neg_a=*/false, av, /*neg_b=*/false, bv,
            /*c_mod=*/(short)0, acc, /*reuse_a=*/false, /*reuse_b=*/false);
      }
    }
  }
  // D layout: lane -> N=lane&15; element e -> M = e + (lane<16 ? 0 : 8)
  if (mrow < 9) {
    float* sp = sim + ((size_t)b * 9 + mrow) * HF * WF + (size_t)y * WF +
                wave * 16 + ((lane < 16) ? 0 : 8);
    #pragma unroll
    for (int e = 0; e < 8; ++e) sp[e] = acc[e];
  }
}

// ---------------- K5: gather + upsample m4 sims -> [B,N,6,128,128] ---------
__global__ __launch_bounds__(256) void output_kernel(const float* __restrict__ sim3,
                                                     const float* __restrict__ sim4,
                                                     float* __restrict__ out) {
  int idx = blockIdx.x * 256 + threadIdx.x;
  constexpr int TOT = Bv * Nv * 6 * H3 * W3;
  if (idx >= TOT) return;
  int x = idx % W3; int r = idx / W3;
  int y = r % H3;  r /= H3;
  int ch = r % 6;  r /= 6;
  int n = r % Nv;  int b = r / Nv;
  int s = ch % 3;
  int f = s * 3 + n;
  float v;
  if (ch < 3) {
    v = sim3[(((size_t)b * 9 + f) * H3 + y) * W3 + x];
  } else {
    const float* sp = sim4 + ((size_t)b * 9 + f) * H4 * W4;
    float ys = fminf(fmaxf((y + 0.5f) * 0.5f - 0.5f, 0.f), (float)(H4 - 1));
    float xs = fminf(fmaxf((x + 0.5f) * 0.5f - 0.5f, 0.f), (float)(W4 - 1));
    int y0 = (int)floorf(ys), x0 = (int)floorf(xs);
    int y1 = min(y0 + 1, H4 - 1), x1 = min(x0 + 1, W4 - 1);
    float wy = ys - y0, wx = xs - x0;
    float ra = sp[y0 * W4 + x0] * (1.f - wy) + sp[y1 * W4 + x0] * wy;
    float rb = sp[y0 * W4 + x1] * (1.f - wy) + sp[y1 * W4 + x1] * wy;
    v = ra * (1.f - wx) + rb * wx;
  }
  out[idx] = v;
}

// ---------------- host-side orchestration ----------------------------------
extern "C" void kernel_launch(void* const* d_in, const int* in_sizes, int n_in,
                              void* d_out, int out_size, void* d_ws, size_t ws_size,
                              hipStream_t stream) {
  (void)in_sizes; (void)n_in; (void)out_size; (void)ws_size;
  const float* map3 = (const float*)d_in[0];
  const float* map4 = (const float*)d_in[1];
  const int*   tlbr = (const int*)d_in[2];
  const int*   ihp  = (const int*)d_in[3];
  const int*   iwp  = (const int*)d_in[4];
  float* out = (float*)d_out;
  char* ws = (char*)d_ws;

  int*    meta = (int*)(ws + OFF_META);
  __bf16* f3   = (__bf16*)(ws + OFF_F3);
  __bf16* f4   = (__bf16*)(ws + OFF_F4);
  __bf16* p3   = (__bf16*)(ws + OFF_P3);
  __bf16* p4   = (__bf16*)(ws + OFF_P4);
  __bf16* fl3  = (__bf16*)(ws + OFF_FL3);
  __bf16* fl4  = (__bf16*)(ws + OFF_FL4);
  float*  s3   = (float*)(ws + OFF_S3);
  float*  s4   = (float*)(ws + OFF_S4);

  meta_kernel<<<1, 64, 0, stream>>>(tlbr, ihp, iwp, meta);

  nhwc_kernel<C3, H3 * W3><<<dim3(H3 * W3 / 32, C3 / 32, Bv), dim3(32, 32), 0, stream>>>(map3, f3);
  nhwc_kernel<C4, H4 * W4><<<dim3(H4 * W4 / 32, C4 / 32, Bv), dim3(32, 32), 0, stream>>>(map4, f4);

  patches_kernel<C3, H3, W3, MAXP3, 0><<<Bv * Nv * MAXP3 * MAXP3, 256, 0, stream>>>(meta, f3, p3);
  patches_kernel<C4, H4, W4, MAXP4, 1><<<Bv * Nv * MAXP4 * MAXP4, 256, 0, stream>>>(meta, f4, p4);

  filters_kernel<C3, MAXP3, MAXF3, 0><<<Bv * MAXF3 * MAXF3 * 16, 256, 0, stream>>>(meta, p3, fl3);
  filters_kernel<C4, MAXP4, MAXF4, 1><<<Bv * MAXF4 * MAXF4 * 16, 256, 0, stream>>>(meta, p4, fl4);

  // one block per output row: 8 waves x 16 px = 128 (m3), 4 waves x 16 = 64 (m4)
  conv_wmma_kernel<C3, H3, W3, MAXF3, 0, 8><<<dim3(H3, Bv), 256, 0, stream>>>(meta, f3, fl3, s3);
  conv_wmma_kernel<C4, H4, W4, MAXF4, 1, 4><<<dim3(H4, Bv), 128, 0, stream>>>(meta, f4, fl4, s4);

  output_kernel<<<(Bv * Nv * 6 * H3 * W3 + 255) / 256, 256, 0, stream>>>(s3, s4, out);
}
